// HomeoProjRepairModule_3375844295068
// MI455X (gfx1250) — matrix operand order repaired and checked by reading, not compiled, not run on hardware
//
#include <hip/hip_runtime.h>
#include <hip/hip_bf16.h>

typedef _Float16 f16;
typedef __attribute__((ext_vector_type(8)))  f16   v8h;
typedef __attribute__((ext_vector_type(16))) f16   v16h;
typedef __attribute__((ext_vector_type(8)))  float v8f;

#define NL    3
#define N_    96
#define HID_  48
#define HIDP  64     // HID padded to K multiple of 32
#define T_    64
#define B_    2048
#define YD    192
#define NO_   96
#define NI_   96
#define NSTEP 8
#define TILE  16     // batch rows per workgroup (one 16-row WMMA tile)
#define BLOCK 32     // one wave32 per workgroup -> barriers become s_nop

// ---- workspace layout (f16 element offsets) -------------------------------
#define WJ_OFF   0                                   // NL x 48 x 96   masked mj_W
#define W1_OFF   (WJ_OFF   + NL*HID_*N_)             // NL x 48 x 64   masked m1_W (K-padded)
#define W2_OFF   (W1_OFF   + NL*HID_*HIDP)           // NL x 192 x 64  masked m2_W (K-padded)
#define WCND_OFF (W2_OFF   + NL*YD*HIDP)             // NL x 48 x 64   mj_Wc
#define LUF_OFF  (WCND_OFF + NL*HID_*T_)             // NL x 96 x 96   lu_mat^T
#define LUI_OFF  (LUF_OFF  + NL*N_*N_)               // NL x 96 x 96   inv(lu_mat)^T
#define WCT_OFF  (LUI_OFF  + NL*N_*N_)               // 96 x 64        Wc^T
#define WPT_OFF  (WCT_OFF  + NO_*T_)                 // 96 x 96        Wp^T
#define GW_OFF   (WPT_OFF  + NO_*N_)                 // 96 x 192       G
#define HXW_OFF  (GW_OFF   + NI_*YD)                 // 96 x 64        Hx

struct Params {
  const float *f, *x, *an1_w, *an1_b, *an2_w, *an2_b, *anF_w, *anF_b;
  const float *lu_P, *lu_L, *lu_U, *lu_logS, *lu_signS;
  const float *mj_W, *mj_b, *mj_Wc, *m1_W, *m1_b, *m2_W, *m2_b;
  const float *Lb, *Ub, *Wc, *Wp, *bc, *G, *Hx, *h;
  f16*  ws;
  float* out;
};

// ---- WMMA fragment loaders (ISA 7.12.2 layouts) ---------------------------
__device__ __forceinline__ v16h cat16(v8h lo, v8h hi) {
  return __builtin_shufflevector(lo, hi, 0,1,2,3,4,5,6,7,8,9,10,11,12,13,14,15);
}
// A: 16x32 f16, row-major LDS, row stride YD halves. lane m = lane&15,
// lanes<16 hold K 0-7 & 16-23, lanes>=16 hold K 8-15 & 24-31 -> 2x ds_read_b128
__device__ __forceinline__ v16h load_a_frag(const f16* AH0, int k0, int lane) {
  const f16* p = AH0 + (lane & 15) * YD + k0 + ((lane >> 4) << 3);
  return cat16(*(const v8h*)p, *(const v8h*)(p + 16));
}
// B: 32x16 (KxN), stored out-major [n][k] in global. lane n = lane&15,
// lanes<16 hold K 0-7 & 16-23, lanes>=16 hold K 8-15 & 24-31 -> 2x global_load_b128
__device__ __forceinline__ v16h load_b_frag(const f16* W, int kstride, int n0, int k0, int lane) {
  const f16* p = W + (n0 + (lane & 15)) * kstride + k0 + ((lane >> 4) << 3);
  return cat16(*(const v8h*)p, *(const v8h*)(p + 16));
}

// C = A(16 x K from LDS) * B(K x NCOLS from global), f32 accum -> LDS out.
// A fragments are hoisted: loaded once from LDS, reused across all col tiles.
template<int NCOLS, int K, int KSTRIDE>
__device__ __forceinline__ void gemm_tile(const f16* AH0, const f16* W,
                                          float* OUT, int ostride, int lane) {
  v16h afrag[K / 32];
  #pragma unroll
  for (int k0 = 0; k0 < K; k0 += 32) afrag[k0 / 32] = load_a_frag(AH0, k0, lane);
  #pragma unroll
  for (int c0 = 0; c0 < NCOLS; c0 += 16) {
    v8f acc = {0.f,0.f,0.f,0.f,0.f,0.f,0.f,0.f};
    #pragma unroll
    for (int k0 = 0; k0 < K; k0 += 32) {
      v16h b = load_b_frag(W, KSTRIDE, c0, k0, lane);
      acc = __builtin_amdgcn_wmma_f32_16x16x32_f16(false, afrag[k0 / 32], false, b,
                                                   (short)0, acc, false, false);
    }
    int col = c0 + (lane & 15);
    int rb  = (lane >> 4) << 3;           // lanes 0-15 -> rows 0-7, 16-31 -> rows 8-15
    #pragma unroll
    for (int r = 0; r < 8; ++r) OUT[(rb + r) * ostride + col] = acc[r];
  }
}

// ---- prep: mask/transpose/f16-convert weights; Gauss-Jordan LU inverses ---
__global__ __launch_bounds__(128) void prep_kernel(Params P) {
  __shared__ float Ms[N_*N_];
  __shared__ float Vs[N_*N_];
  __shared__ float fac[N_];
  __shared__ float pivinv;
  __shared__ int   perm[N_];
  const int tid = threadIdx.x;
  f16* ws = P.ws;

  if (blockIdx.x < NL) {
    const int l = blockIdx.x;
    const float* luL = P.lu_L + l*N_*N_;
    const float* luU = P.lu_U + l*N_*N_;
    const float* luP = P.lu_P + l*N_*N_;
    // Vs = (L*LMASK+I) @ (U*UMASK+diag(sign*exp(logS)))
    for (int idx = tid; idx < N_*N_; idx += 128) {
      int i = idx / N_, j = idx % N_;
      float s = 0.f;
      int kmax = i < j ? i : j;
      for (int k = 0; k <= kmax; ++k) {
        float lv = (k < i) ? luL[i*N_ + k] : 1.0f;
        float uv = (k < j) ? luU[k*N_ + j]
                           : P.lu_signS[l*N_ + j] * __expf(P.lu_logS[l*N_ + j]);
        s += lv * uv;
      }
      Vs[idx] = s;
    }
    if (tid < N_) {
      int pi = 0;
      for (int j = 0; j < N_; ++j) if (luP[tid*N_ + j] > 0.5f) pi = j;
      perm[tid] = pi;
    }
    __syncthreads();
    // M = P @ LU ; emit forward matrix transposed ([out][in])
    for (int idx = tid; idx < N_*N_; idx += 128) {
      int i = idx / N_, j = idx % N_;
      float v = Vs[perm[i]*N_ + j];
      Ms[idx] = v;
      ws[LUF_OFF + l*N_*N_ + j*N_ + i] = (f16)v;
    }
    __syncthreads();
    for (int idx = tid; idx < N_*N_; idx += 128)
      Vs[idx] = (idx / N_ == idx % N_) ? 1.0f : 0.0f;
    __syncthreads();
    // Gauss-Jordan inversion
    for (int k = 0; k < N_; ++k) {
      if (tid == 0) pivinv = 1.0f / Ms[k*N_ + k];
      __syncthreads();
      for (int j = tid; j < N_; j += 128) { Ms[k*N_ + j] *= pivinv; Vs[k*N_ + j] *= pivinv; }
      __syncthreads();
      if (tid < N_) fac[tid] = (tid == k) ? 0.0f : Ms[tid*N_ + k];
      __syncthreads();
      for (int idx = tid; idx < N_*N_; idx += 128) {
        int i = idx / N_, j = idx % N_;
        if (i != k) {
          float fk = fac[i];
          Ms[idx] -= fk * Ms[k*N_ + j];
          Vs[idx] -= fk * Vs[k*N_ + j];
        }
      }
      __syncthreads();
    }
    for (int idx = tid; idx < N_*N_; idx += 128) {
      int i = idx / N_, j = idx % N_;
      ws[LUI_OFF + l*N_*N_ + j*N_ + i] = (f16)Vs[idx];
    }
  } else {
    // masked / transposed / padded f16 weight conversion
    for (int idx = tid; idx < NL*HID_*N_; idx += 128) {
      int n = idx % N_, hh = (idx / N_) % HID_;
      ws[WJ_OFF + idx] = (f16)(P.mj_W[idx] * ((hh >= n) ? 1.0f : 0.0f));
    }
    for (int idx = tid; idx < NL*HID_*HIDP; idx += 128) {
      int j = idx % HIDP, i = (idx / HIDP) % HID_, l = idx / (HID_*HIDP);
      float v = (j < HID_ && i >= j) ? P.m1_W[(l*HID_ + i)*HID_ + j] : 0.0f;
      ws[W1_OFF + idx] = (f16)v;
    }
    for (int idx = tid; idx < NL*YD*HIDP; idx += 128) {
      int j = idx % HIDP, o = (idx / HIDP) % YD, l = idx / (YD*HIDP);
      float v = 0.0f;
      if (j < HID_ && ((o % N_) - 1) >= j) v = P.m2_W[(l*YD + o)*HID_ + j];
      ws[W2_OFF + idx] = (f16)v;
    }
    for (int idx = tid; idx < NL*HID_*T_; idx += 128)
      ws[WCND_OFF + idx] = (f16)P.mj_Wc[idx];
    for (int idx = tid; idx < NO_*T_; idx += 128) {
      int t = idx % T_, no = idx / T_;
      ws[WCT_OFF + idx] = (f16)P.Wc[t*NO_ + no];
    }
    for (int idx = tid; idx < NO_*N_; idx += 128) {
      int n = idx % N_, no = idx / N_;
      ws[WPT_OFF + idx] = (f16)P.Wp[n*NO_ + no];
    }
    for (int idx = tid; idx < NI_*YD; idx += 128)
      ws[GW_OFF + idx] = (f16)P.G[idx];
    for (int idx = tid; idx < NI_*T_; idx += 128)
      ws[HXW_OFF + idx] = (f16)P.Hx[idx];
  }
}

// ---- main flow kernel: one 16-row batch tile per single-wave workgroup ----
__global__ __launch_bounds__(BLOCK) void flow_kernel(Params P) {
  __shared__ __align__(16) f16 AH[TILE*YD];   // f16 A-fragment staging
  __shared__ float OB[TILE*YD];               // f32 GEMM output / scratch
  __shared__ float Z[TILE*N_];                // current activation (f32 master)
  __shared__ float XL[TILE*N_];               // x_lat
  __shared__ float CND[NL][TILE*HID_];        // cond + mj_b per layer
  __shared__ float RX[TILE*NI_];              // x@Hx^T + h
  __shared__ float XWC[TILE*NO_];             // x@Wc
  __shared__ float PRE[TILE*HID_];            // incremental layer-1 preact
  __shared__ float alo[TILE], aup[TILE], alp[TILE], tm[TILE], ta[TILE], xis[TILE];

  const int tid  = threadIdx.x;
  const int lane = tid & 31;
  const int r0   = blockIdx.x * TILE;
  const f16* ws  = P.ws;

  // ---- stage x tile, precompute cond / rx / xwc ----
  for (int idx = tid; idx < TILE*T_; idx += BLOCK) {
    int r = idx / T_, c = idx % T_;
    AH[r*YD + c] = (f16)P.x[(r0 + r)*T_ + c];
  }
  __syncthreads();
  for (int l = 0; l < NL; ++l)
    gemm_tile<HID_, T_, T_>(AH, ws + WCND_OFF + l*HID_*T_, CND[l], HID_, lane);
  gemm_tile<NI_, T_, T_>(AH, ws + HXW_OFF, RX, NI_, lane);
  gemm_tile<NO_, T_, T_>(AH, ws + WCT_OFF, XWC, NO_, lane);
  __syncthreads();
  for (int idx = tid; idx < TILE*HID_; idx += BLOCK) {
    int c = idx % HID_;
    for (int l = 0; l < NL; ++l) CND[l][idx] += P.mj_b[l*HID_ + c];
  }
  for (int idx = tid; idx < TILE*NI_; idx += BLOCK) RX[idx] += P.h[idx % NI_];
  // z0 = logit((f-LO)/SCALE)*exp(-anF_w)+anF_b
  for (int idx = tid; idx < TILE*N_; idx += BLOCK) {
    int r = idx / N_, c = idx % N_;
    float xx = (P.f[(r0 + r)*N_ + c] + 0.1f) * (1.0f / 1.2f);
    float z  = __logf(xx / (1.0f - xx));
    Z[idx] = z * __expf(-P.anF_w[c]) + P.anF_b[c];
  }
  __syncthreads();

  // ---- inn_inv: autoregressive inverse scan per layer (reversed) ----
  for (int l = NL - 1; l >= 0; --l) {
    for (int idx = tid; idx < TILE*HID_; idx += BLOCK) PRE[idx] = CND[l][idx];
    for (int idx = tid; idx < TILE*N_;   idx += BLOCK) OB[(idx/N_)*YD + 96 + (idx % N_)] = 0.0f;
    for (int idx = tid; idx < TILE*16;   idx += BLOCK) AH[(idx/16)*YD + 48 + (idx % 16)] = (f16)0.0f;
    __syncthreads();
    const f16* W1l = ws + W1_OFF + l*HID_*HIDP;
    const f16* W2l = ws + W2_OFF + l*YD*HIDP;
    const f16* Wjl = ws + WJ_OFF + l*HID_*N_;
    for (int i = 0; i < N_; ++i) {
      for (int idx = tid; idx < TILE*HID_; idx += BLOCK) {
        float v = PRE[idx];
        AH[(idx/HID_)*YD + (idx % HID_)] = (f16)(v > 0.f ? v : 0.f);
      }
      __syncthreads();
      gemm_tile<HID_, HIDP, HIDP>(AH, W1l, OB, YD, lane);   // hidden2 pre-act
      __syncthreads();
      { // only the 2 needed output columns: scalar dots over HID
        int r = tid & 15, which = tid >> 4;
        int col = which ? (96 + i) : i;
        float dot = P.m2_b[l*YD + col];
        #pragma unroll 8
        for (int j = 0; j < HID_; ++j) {
          float hv = OB[r*YD + j] + P.m1_b[l*HID_ + j];
          hv = hv > 0.f ? hv : 0.f;
          dot += hv * (float)W2l[col*HIDP + j];
        }
        if (which) ta[r] = dot; else tm[r] = dot;
      }
      __syncthreads();
      if (tid < TILE) {
        float xi = Z[tid*N_ + i] * __expf(ta[tid]) + tm[tid];
        xis[tid] = xi;
        OB[tid*YD + 96 + i] = xi;      // xc column i
      }
      __syncthreads();
      for (int idx = tid; idx < TILE*HID_; idx += BLOCK) {  // rank-1 PRE update
        int r = idx / HID_, j = idx % HID_;
        PRE[idx] += xis[r] * (float)Wjl[j*N_ + i];
      }
      __syncthreads();
    }
    // z := xc ; inverse affine an2 ; z @ inv(lu) ; inverse affine an1
    for (int idx = tid; idx < TILE*N_; idx += BLOCK) {
      int r = idx / N_, c = idx % N_;
      float z = OB[r*YD + 96 + c] * __expf(-P.an2_w[l*N_ + c]) + P.an2_b[l*N_ + c];
      Z[idx] = z; AH[r*YD + c] = (f16)z;
    }
    __syncthreads();
    gemm_tile<N_, N_, N_>(AH, ws + LUI_OFF + l*N_*N_, OB, YD, lane);
    __syncthreads();
    for (int idx = tid; idx < TILE*N_; idx += BLOCK) {
      int r = idx / N_, c = idx % N_;
      Z[idx] = OB[r*YD + c] * __expf(-P.an1_w[l*N_ + c]) + P.an1_b[l*N_ + c];
    }
    __syncthreads();
  }
  for (int idx = tid; idx < TILE*N_; idx += BLOCK) XL[idx] = Z[idx];
  if (tid < TILE) { alo[tid] = 0.0f; aup[tid] = 1.0f; }
  __syncthreads();

  // ---- bisection (8 iters) + final forward pass ----
  for (int it = 0; it <= NSTEP; ++it) {
    const bool fin = (it == NSTEP);
    if (tid < TILE) alp[tid] = fin ? alo[tid] : 0.5f * (alo[tid] + aup[tid]);
    __syncthreads();
    for (int idx = tid; idx < TILE*N_; idx += BLOCK) {
      int r = idx / N_, c = idx % N_;
      float z = alp[r] * (XL[idx] - 0.5f) + 0.5f;
      Z[idx] = z; AH[r*YD + c] = (f16)z;
    }
    __syncthreads();
    // inn_fwd
    for (int l = 0; l < NL; ++l) {
      for (int idx = tid; idx < TILE*N_; idx += BLOCK) {
        int r = idx / N_, c = idx % N_;
        float z = (Z[idx] - P.an1_b[l*N_ + c]) * __expf(P.an1_w[l*N_ + c]);
        Z[idx] = z; AH[r*YD + c] = (f16)z;
      }
      __syncthreads();
      gemm_tile<N_, N_, N_>(AH, ws + LUF_OFF + l*N_*N_, OB, YD, lane);
      __syncthreads();
      for (int idx = tid; idx < TILE*N_; idx += BLOCK) {
        int r = idx / N_, c = idx % N_;
        float z = (OB[r*YD + c] - P.an2_b[l*N_ + c]) * __expf(P.an2_w[l*N_ + c]);
        Z[idx] = z; AH[r*YD + c] = (f16)z;
      }
      __syncthreads();
      gemm_tile<HID_, N_, N_>(AH, ws + WJ_OFF + l*HID_*N_, OB, YD, lane);
      __syncthreads();
      for (int idx = tid; idx < TILE*HID_; idx += BLOCK) {
        int r = idx / HID_, j = idx % HID_;
        float v = OB[r*YD + j] + CND[l][idx];
        AH[r*YD + j] = (f16)(v > 0.f ? v : 0.f);
      }
      for (int idx = tid; idx < TILE*16; idx += BLOCK)
        AH[(idx/16)*YD + 48 + (idx % 16)] = (f16)0.0f;    // K-pad
      __syncthreads();
      gemm_tile<HID_, HIDP, HIDP>(AH, ws + W1_OFF + l*HID_*HIDP, OB, YD, lane);
      __syncthreads();
      for (int idx = tid; idx < TILE*HID_; idx += BLOCK) {
        int r = idx / HID_, j = idx % HID_;
        float v = OB[r*YD + j] + P.m1_b[l*HID_ + j];
        AH[r*YD + j] = (f16)(v > 0.f ? v : 0.f);
      }
      __syncthreads();
      gemm_tile<YD, HIDP, HIDP>(AH, ws + W2_OFF + l*YD*HIDP, OB, YD, lane);
      __syncthreads();
      for (int idx = tid; idx < TILE*N_; idx += BLOCK) {
        int r = idx / N_, c = idx % N_;
        float m = OB[r*YD + c]      + P.m2_b[l*YD + c];
        float a = OB[r*YD + 96 + c] + P.m2_b[l*YD + 96 + c];
        float z = (Z[idx] - m) * __expf(-a);
        Z[idx] = z; AH[r*YD + c] = (f16)z;
      }
      __syncthreads();
    }
    // final affine + sigmoid + y, then complete()
    for (int idx = tid; idx < TILE*N_; idx += BLOCK) {
      int r = idx / N_, c = idx % N_;
      float z  = (Z[idx] - P.anF_b[c]) * __expf(P.anF_w[c]);
      float xt = 1.2f / (1.0f + __expf(-z)) - 0.1f;
      float y  = xt * (P.Ub[c] - P.Lb[c]) + P.Lb[c];
      Z[idx] = y; AH[r*YD + c] = (f16)y;
    }
    __syncthreads();
    gemm_tile<NO_, N_, N_>(AH, ws + WPT_OFF, OB + 96, YD, lane);  // yp@Wp -> cols 96..191
    __syncthreads();
    for (int idx = tid; idx < TILE*NO_; idx += BLOCK) {
      int r = idx / NO_, c = idx % NO_;
      float o = OB[r*YD + 96 + c] + XWC[idx] + P.bc[c];
      OB[r*YD + 96 + c] = o;
      AH[r*YD + 96 + c] = (f16)o;        // y_full f16 (cols 0..95 already y)
    }
    __syncthreads();
    if (!fin) {
      gemm_tile<NI_, YD, YD>(AH, ws + GW_OFF, OB, YD, lane);      // y_full @ G^T
      __syncthreads();
      if (tid < TILE) {
        float p = 0.0f;
        for (int c = 0; c < NI_; ++c) {
          float v = OB[tid*YD + c] - RX[tid*NI_ + c];
          if (v > p) p = v;
        }
        if (p < 1e-4f) alo[tid] = alp[tid]; else aup[tid] = alp[tid];
      }
      __syncthreads();
    } else {
      for (int idx = tid; idx < TILE*YD; idx += BLOCK) {
        int r = idx / YD, c = idx % YD;
        P.out[(r0 + r)*YD + c] = (c < N_) ? Z[r*N_ + c] : OB[r*YD + c];
      }
    }
  }
}

extern "C" void kernel_launch(void* const* d_in, const int* in_sizes, int n_in,
                              void* d_out, int out_size, void* d_ws, size_t ws_size,
                              hipStream_t stream) {
  (void)in_sizes; (void)n_in; (void)out_size; (void)ws_size;
  Params P;
  P.f       = (const float*)d_in[0];
  P.x       = (const float*)d_in[1];
  P.an1_w   = (const float*)d_in[2];
  P.an1_b   = (const float*)d_in[3];
  P.an2_w   = (const float*)d_in[4];
  P.an2_b   = (const float*)d_in[5];
  P.anF_w   = (const float*)d_in[6];
  P.anF_b   = (const float*)d_in[7];
  P.lu_P    = (const float*)d_in[8];
  P.lu_L    = (const float*)d_in[9];
  P.lu_U    = (const float*)d_in[10];
  P.lu_logS = (const float*)d_in[11];
  P.lu_signS= (const float*)d_in[12];
  P.mj_W    = (const float*)d_in[13];
  P.mj_b    = (const float*)d_in[14];
  P.mj_Wc   = (const float*)d_in[15];
  P.m1_W    = (const float*)d_in[16];
  P.m1_b    = (const float*)d_in[17];
  P.m2_W    = (const float*)d_in[18];
  P.m2_b    = (const float*)d_in[19];
  P.Lb      = (const float*)d_in[20];
  P.Ub      = (const float*)d_in[21];
  P.Wc      = (const float*)d_in[22];
  P.Wp      = (const float*)d_in[23];
  P.bc      = (const float*)d_in[24];
  P.G       = (const float*)d_in[25];
  P.Hx      = (const float*)d_in[26];
  P.h       = (const float*)d_in[27];
  P.ws      = (f16*)d_ws;
  P.out     = (float*)d_out;

  prep_kernel<<<dim3(NL + 1), dim3(128), 0, stream>>>(P);
  flow_kernel<<<dim3(B_ / TILE), dim3(BLOCK), 0, stream>>>(P);
}
